// SupCon_2783138808531
// MI455X (gfx1250) — matrix operand order repaired and checked by reading, not compiled, not run on hardware
//
#include <hip/hip_runtime.h>
#include <stdint.h>

// ---------------------------------------------------------------------------
// SupCon loss, fused, for MI455X (gfx1250, wave32, WMMA + async global->LDS).
//   sim = (z/||z||) @ (z/||z||)^T / T is never materialized (2 MB f16 operand
//   is L2-resident). Each 128-thread block owns 64 rows (4 waves x 16 rows),
//   double-buffers 16-column B tiles in LDS via global_load_async_to_lds_b128,
//   and runs v_wmma_f32_16x16x32_f16 (K=128 -> 4 WMMA per tile) with a fully
//   branchless streaming softmax/positive-mask epilogue.
// ---------------------------------------------------------------------------

typedef __attribute__((ext_vector_type(16))) _Float16 v16h;
typedef __attribute__((ext_vector_type(8)))  _Float16 h8;
typedef __attribute__((ext_vector_type(4)))  _Float16 h4;
typedef __attribute__((ext_vector_type(8)))  float    v8f;

#define DIM 128
#define BROW 136                           // padded LDS row (halves): 68 words -> conflict-free
#define SQRT_INV_T 3.16227766016837933f    // sqrt(1/0.1) folded into both operands
#define SHIFT 10.0f                        // max |sim/T| -> exact LSE shift

// --- Kernel 1: zh[row] = f16( z[row] * sqrt(1/T) / max(||z[row]||, eps) ) ---
__global__ void __launch_bounds__(256) supcon_normalize(
    const float* __restrict__ z, _Float16* __restrict__ zh, int n) {
  const int lane = threadIdx.x & 31;
  const int row  = (int)((blockIdx.x * blockDim.x + threadIdx.x) >> 5);
  if (row >= n) return;
  const float4 v = ((const float4*)(z + (size_t)row * DIM))[lane];  // 32*4 = 128
  float ss = v.x * v.x + v.y * v.y + v.z * v.z + v.w * v.w;
#pragma unroll
  for (int m = 16; m >= 1; m >>= 1) ss += __shfl_xor(ss, m, 32);
  const float nrm = fmaxf(sqrtf(ss), 1e-8f);
  const float sc  = SQRT_INV_T / nrm;
  h4 h;
  h[0] = (_Float16)(v.x * sc);
  h[1] = (_Float16)(v.y * sc);
  h[2] = (_Float16)(v.z * sc);
  h[3] = (_Float16)(v.w * sc);
  ((h4*)(zh + (size_t)row * DIM))[lane] = h;
}

// Stage one 16-col x 128-half B tile (4 KB) into LDS with async copies.
// 128 threads: thread tid copies 32 B of column (tid>>3) at chunk (tid&7).
__device__ __forceinline__ void stage_tile(const _Float16* __restrict__ zh,
                                           uint32_t lds_byte, int tile, int tid) {
  const int col = tid >> 3;                 // 0..15
  const int ch  = tid & 7;                  // 0..7, each = 2 x 16 B
  const _Float16* g = zh + (size_t)(tile * 16 + col) * DIM + ch * 16;
  const uint32_t l = lds_byte + (uint32_t)(col * (BROW * 2) + ch * 32);
  // INST_OFFSET adds to BOTH the LDS and global addresses (ISA 08_async_tensor).
  asm volatile("global_load_async_to_lds_b128 %0, %1, off\n\t"
               "global_load_async_to_lds_b128 %0, %1, off offset:32"
               :: "v"(l), "v"(g) : "memory");
}

// --- Kernel 2: fused GEMM + masked log-softmax + supervised-contrastive loss -
__global__ void __launch_bounds__(128) supcon_main(
    const _Float16* __restrict__ zh, const int* __restrict__ y,
    float* __restrict__ out, int n) {
  __shared__ __align__(32) _Float16 smem[2][16][BROW];   // 2 x 4.25 KB tiles

  const int tid  = threadIdx.x;
  const int lane = tid & 31;
  const int wave = tid >> 5;
  const int li   = lane & 15;       // column-within-tile / row-within-block
  const int hi   = lane >> 4;       // half-wave select per ISA WMMA layouts
  const int rb   = ((int)blockIdx.x * 4 + wave) * 16;    // this wave's 16 rows
  const int rbh  = rb + hi * 8;     // first row covered by this lane's C VGPRs

  // A fragments: rows rb..rb+15, K=0..127 in 4 chunks of 32, resident in VGPRs.
  // ISA 16-bit A 16x32 layout: lane<16 holds K {c*32+0..7, c*32+16..23},
  //                            lane>=16 holds K {c*32+8..15, c*32+24..31}.
  v16h a[4];
  {
    const _Float16* arow = zh + (size_t)(rb + li) * DIM;
#pragma unroll
    for (int c = 0; c < 4; ++c) {
      const h8 lo = *(const h8*)(arow + c * 32 + hi * 8);
      const h8 hh = *(const h8*)(arow + c * 32 + 16 + hi * 8);
      v16h t;
#pragma unroll
      for (int i = 0; i < 8; ++i) { t[i] = lo[i]; t[i + 8] = hh[i]; }
      a[c] = t;
    }
  }

  int yrow[8];
#pragma unroll
  for (int r = 0; r < 8; ++r) yrow[r] = y[rbh + r];

  float s[8], ps[8];
  int   pc[8];
#pragma unroll
  for (int r = 0; r < 8; ++r) { s[r] = 0.0f; ps[r] = 0.0f; pc[r] = 0; }

  const uint32_t sbase = (uint32_t)(size_t)&smem[0][0][0];  // wave-relative LDS byte addr
  const int ntiles = n >> 4;

  stage_tile(zh, sbase, 0, tid);                            // prologue: tile 0 -> buf 0

  for (int t = 0; t < ntiles; ++t) {
    // Stage tile t+1 into the other buffer (wraps at the end; harmless).
    const int tn = (t + 1) & (ntiles - 1);
    stage_tile(zh, sbase + (uint32_t)(((t + 1) & 1) * 16 * BROW * 2), tn, tid);

    // Prefetch tile t+2's global data into L2 (global_prefetch_b8).
    __builtin_prefetch(zh + (size_t)((((t + 2) & (ntiles - 1)) * 16 + (tid >> 3)) * DIM), 0, 1);

    // Wait for tile t's async copies (only tile t+1's 2 remain), then publish.
    asm volatile("s_wait_asynccnt 0x2" ::: "memory");
    __syncthreads();

    // B fragments from LDS: lane<16 = K {0..7, 16..23} of column li, lane>=16 = K {8..15, 24..31}.
    const _Float16* srow = &smem[t & 1][li][0];
    v16h b[4];
#pragma unroll
    for (int c = 0; c < 4; ++c) {
      const h8 lo = *(const h8*)(srow + c * 32 + hi * 8);
      const h8 hh = *(const h8*)(srow + c * 32 + 16 + hi * 8);
      v16h tb;
#pragma unroll
      for (int i = 0; i < 8; ++i) { tb[i] = lo[i]; tb[i + 8] = hh[i]; }
      b[c] = tb;
    }

    v8f acc = {};
#pragma unroll
    for (int c = 0; c < 4; ++c)
      acc = __builtin_amdgcn_wmma_f32_16x16x32_f16(
          false, a[c], false, b[c], (short)0, acc, false, false);

    const int col = (t << 4) + li;            // this lane's column
    const int yc  = y[col];
    const int dg  = col - rbh;                // diagonal when dg == r

    // Branchless streaming update (v_cndmask, no EXEC branches).
#pragma unroll
    for (int r = 0; r < 8; ++r) {
      const float v     = acc[r];             // = sim/T (f32 accumulation)
      const bool  ndiag = (dg != r);
      const bool  pos   = ndiag && (yc == yrow[r]);
      s[r]  += ndiag ? __expf(v - SHIFT) : 0.0f;
      ps[r] += pos ? v : 0.0f;
      pc[r] += pos ? 1 : 0;
    }

    __syncthreads();   // everyone done with buf (t&1) before it is restaged
  }

  // Row-wise reduction across the 16 lanes holding each row's columns.
#pragma unroll
  for (int r = 0; r < 8; ++r) {
#pragma unroll
    for (int m = 1; m <= 8; m <<= 1) {        // stays within 16-lane halves
      s[r]  += __shfl_xor(s[r],  m, 32);
      ps[r] += __shfl_xor(ps[r], m, 32);
      pc[r] += __shfl_xor(pc[r], m, 32);
    }
  }

  if (li == 0) {     // lane 0 -> rows rbh..rbh+7, lane 16 -> rbh..rbh+7 (hi half)
#pragma unroll
    for (int r = 0; r < 8; ++r) {
      const float lse  = __logf(s[r]) + SHIFT;
      const float loss = (pc[r] > 0) ? (lse - ps[r] / (float)pc[r]) : 0.0f;
      out[rbh + r] = loss;
    }
  }
}

extern "C" void kernel_launch(void* const* d_in, const int* in_sizes, int n_in,
                              void* d_out, int out_size, void* d_ws, size_t ws_size,
                              hipStream_t stream) {
  const float* z = (const float*)d_in[0];
  const int*   y = (const int*)d_in[1];
  float*       out = (float*)d_out;
  const int n = in_sizes[1];                  // 8192
  _Float16* zh = (_Float16*)d_ws;             // n*128*2 B = 2 MB scratch

  supcon_normalize<<<(n + 7) / 8, 256, 0, stream>>>(z, zh, n);
  supcon_main<<<n / 64, 128, 0, stream>>>(zh, y, out, n);
}
